// Perceiver_54408645706016
// MI455X (gfx1250) — matrix operand order repaired and checked by reading, not compiled, not run on hardware
//
#include <hip/hip_runtime.h>
#include <hip/hip_bf16.h>
#include <math.h>

typedef __bf16 bf16;
typedef __attribute__((ext_vector_type(16))) __bf16 v16bf;
typedef __attribute__((ext_vector_type(8)))  float  v8f;
typedef int v4i __attribute__((vector_size(16)));

#define LEN_X   512
#define EMB     512
#define DMODEL  1024
#define SLAT    64
#define TSTEP   32
#define NLAYER  8
#define NHEAD   8
#define HD      128
#define FFDIM   4096
#define NTOK    2560      // LEN_X + TSTEP*SLAT
#define NLAT    2048      // latent rows (carry state between layers)
#define OUTDIM  1024

#if defined(__HIP_DEVICE_COMPILE__) &&                                         \
    __has_builtin(__builtin_amdgcn_global_load_async_to_lds_b128) &&           \
    __has_builtin(__builtin_amdgcn_s_wait_asynccnt)
#define USE_ASYNC_LDS 1
#else
#define USE_ASYNC_LDS 0
#endif

// ---------------------------------------------------------------------------
// Async global -> LDS copy of 16 bytes (ASYNCcnt-tracked on gfx1250).
// ---------------------------------------------------------------------------
static __device__ __forceinline__ void stage16(const bf16* g, bf16* l) {
#if USE_ASYNC_LDS
  __builtin_amdgcn_global_load_async_to_lds_b128(
      (__attribute__((address_space(1))) v4i*)(g),
      (__attribute__((address_space(3))) v4i*)(l), 0, 0);
#else
  *(uint4*)l = *(const uint4*)g;
#endif
}

static __device__ __forceinline__ void wait_async_all() {
#if USE_ASYNC_LDS
  __builtin_amdgcn_s_wait_asynccnt(0);
#endif
}

// ---------------------------------------------------------------------------
// WMMA fragment helpers (layouts per CDNA5 ISA 7.12.2)
// ---------------------------------------------------------------------------
static __device__ __forceinline__ v8f zero8() {
  v8f z;
#pragma unroll
  for (int i = 0; i < 8; i++) z[i] = 0.0f;
  return z;
}

static __device__ __forceinline__ v8f wmma_bf16(v16bf a, v16bf b, v8f c) {
  return __builtin_amdgcn_wmma_f32_16x16x32_bf16(false, a, false, b, (short)0, c,
                                                 false, false);
}

// A: 16(M) x 32(K) bf16, row-major source with leading dim ld (elements).
static __device__ __forceinline__ v16bf load_a_frag(const bf16* base, int ld) {
  int lane = threadIdx.x & 31;
  int m    = lane & 15;
  int kh   = (lane >> 4) & 1;
  const bf16* r = base + (size_t)m * ld;
  union { uint4 u[2]; v16bf v; } u;
  u.u[0] = *(const uint4*)(r + 8 * kh);        // K = 8*kh .. 8*kh+7
  u.u[1] = *(const uint4*)(r + 16 + 8 * kh);   // K = 16+8*kh .. +7
  return u.v;
}

// B: 32(K) x 16(N) bf16, K x N row-major source with leading dim ld.
static __device__ __forceinline__ v16bf load_b_frag(const bf16* base, int ld) {
  int lane = threadIdx.x & 31;
  int n    = lane & 15;
  int kh   = (lane >> 4) & 1;
  v16bf v;
#pragma unroll
  for (int s = 0; s < 8; s++) {
    v[s]     = base[(size_t)(8 * kh + s) * ld + n];
    v[s + 8] = base[(size_t)(16 + 8 * kh + s) * ld + n];
  }
  return v;
}

// ---------------------------------------------------------------------------
// Input assembly: fourier features, concat, latent tiling, timestep vector.
// Context rows of x are written once and never modified afterwards.
// ---------------------------------------------------------------------------
__global__ void build_x_kernel(const float* __restrict__ data,
                               const int* __restrict__ tsteps,
                               const float* __restrict__ latent,
                               float* __restrict__ x, int* __restrict__ ts) {
  int idx = blockIdx.x * blockDim.x + threadIdx.x;
  if (idx >= NTOK * DMODEL) return;
  int row = idx / DMODEL, col = idx % DMODEL;
  float val;
  if (row < LEN_X) {
    if (col < EMB) {
      int band   = (col < 256) ? col : col - 256;
      float freq = 1.0f + band * (499.0f / 255.0f);   // linspace(1, 500, 256)
      float ang  = 3.14159265358979f * (float)row * freq;
      val = (col < 256) ? sinf(ang) : cosf(ang);
    } else {
      val = data[row * 512 + (col - 512)];
    }
  } else {
    int li = row - LEN_X;
    val = latent[(li % SLAT) * DMODEL + col];
  }
  x[(size_t)row * DMODEL + col] = val;
  if (col == 0) ts[row] = (row < LEN_X) ? tsteps[row] : (row - LEN_X) / SLAT;
}

__global__ void cvt_bf16_kernel(const float* __restrict__ src,
                                bf16* __restrict__ dst, int n) {
  int i = blockIdx.x * 256 + threadIdx.x;
  if (i < n) dst[i] = (bf16)src[i];
}

// ---------------------------------------------------------------------------
// Row LayerNorm: fp32 in -> bf16 out
// ---------------------------------------------------------------------------
__global__ __launch_bounds__(256) void layernorm_kernel(
    const float* __restrict__ x, const float* __restrict__ g,
    const float* __restrict__ b, bf16* __restrict__ out) {
  int row = blockIdx.x;
  const float* xr = x + (size_t)row * DMODEL;
  __shared__ float red[256];
  float s = 0.0f;
  for (int i = threadIdx.x; i < DMODEL; i += 256) s += xr[i];
  red[threadIdx.x] = s;
  __syncthreads();
  for (int st = 128; st > 0; st >>= 1) {
    if (threadIdx.x < st) red[threadIdx.x] += red[threadIdx.x + st];
    __syncthreads();
  }
  float mean = red[0] / DMODEL;
  __syncthreads();
  float v = 0.0f;
  for (int i = threadIdx.x; i < DMODEL; i += 256) {
    float d = xr[i] - mean;
    v += d * d;
  }
  red[threadIdx.x] = v;
  __syncthreads();
  for (int st = 128; st > 0; st >>= 1) {
    if (threadIdx.x < st) red[threadIdx.x] += red[threadIdx.x + st];
    __syncthreads();
  }
  float rstd = rsqrtf(red[0] / DMODEL + 1e-5f);
  for (int i = threadIdx.x; i < DMODEL; i += 256)
    out[(size_t)row * DMODEL + i] = (bf16)((xr[i] - mean) * rstd * g[i] + b[i]);
}

// ---------------------------------------------------------------------------
// Generic WMMA GEMM, double-buffered async B staging.
// Block = 256 threads (8 waves); block tile 256(M) x 128(N).
// Each wave computes 32x128 (two A frags share every B frag, read with 2-byte
// strided LDS loads). One barrier per 32-wide K-step.
// mode 0: outb bf16 = acc
// mode 1: outb bf16 = gelu(acc + bias)
// mode 2: outf fp32 += acc (+ bias)
// ---------------------------------------------------------------------------
__global__ __launch_bounds__(256) void gemm_kernel(
    const bf16* __restrict__ A, const bf16* __restrict__ B,
    const float* __restrict__ bias, bf16* __restrict__ outb,
    float* __restrict__ outf, int M, int N, int K, int mode) {
  __shared__ bf16 Bs[2][32 * 128];
  int n0   = blockIdx.x * 128;
  int m0   = blockIdx.y * 256;
  int tid  = threadIdx.x;
  int wave = tid >> 5;
  int lane = tid & 31;
  int mw   = m0 + wave * 32;
  (void)M;

  v8f acc0[8], acc1[8];
#pragma unroll
  for (int j = 0; j < 8; j++) { acc0[j] = zero8(); acc1[j] = zero8(); }

  int kk_s = tid >> 3;          // 0..31  (K row within tile)
  int cb_s = (tid & 7) * 16;    // 0..112 (N col within tile)
  bf16* dst0 = &Bs[0][kk_s * 128 + cb_s];
  bf16* dst1 = &Bs[1][kk_s * 128 + cb_s];

  const bf16*  bsrc  = B + (size_t)kk_s * N + n0 + cb_s;
  const size_t bstep = (size_t)32 * N;

  stage16(bsrc, dst0);
  stage16(bsrc + 8, dst0 + 8);
  bsrc += bstep;
  wait_async_all();
  __syncthreads();

  const bf16* aptr = A + (size_t)mw * K;
  int cur = 0;
  for (int kt = 0; kt < K; kt += 32) {
    if (kt + 32 < K) {
      bf16* d = cur ? dst0 : dst1;
      stage16(bsrc, d);
      stage16(bsrc + 8, d + 8);
      bsrc += bstep;
    }
    v16bf a0 = load_a_frag(aptr + kt, K);
    v16bf a1 = load_a_frag(aptr + (size_t)16 * K + kt, K);
    const bf16* bb = cur ? &Bs[1][0] : &Bs[0][0];
#pragma unroll
    for (int j = 0; j < 8; j++) {
      v16bf b  = load_b_frag(bb + j * 16, 128);
      acc0[j]  = wmma_bf16(a0, b, acc0[j]);
      acc1[j]  = wmma_bf16(a1, b, acc1[j]);
    }
    wait_async_all();
    __syncthreads();
    cur ^= 1;
  }

  int half  = lane >> 4;
  int col_l = lane & 15;
#pragma unroll
  for (int rg = 0; rg < 2; rg++) {
#pragma unroll
    for (int j = 0; j < 8; j++) {
#pragma unroll
      for (int r = 0; r < 8; r++) {
        int row = mw + rg * 16 + (half ? 8 + r : r);
        int col = n0 + j * 16 + col_l;
        float v = rg ? acc1[j][r] : acc0[j][r];
        if (mode == 0) {
          outb[(size_t)row * N + col] = (bf16)v;
        } else if (mode == 1) {
          v += bias[col];
          float gl = 0.5f * v *
                     (1.0f + tanhf(0.7978845608f * (v + 0.044715f * v * v * v)));
          outb[(size_t)row * N + col] = (bf16)gl;
        } else {
          float bb2 = bias ? bias[col] : 0.0f;
          outf[(size_t)row * N + col] += v + bb2;
        }
      }
    }
  }
}

// K (NTOK x D bf16) -> kT (NHEAD x HD x NTOK bf16)
__global__ void transpose_k_kernel(const bf16* __restrict__ k,
                                   bf16* __restrict__ kT) {
  int idx = blockIdx.x * 256 + threadIdx.x;
  if (idx >= NTOK * DMODEL) return;
  int n = idx / DMODEL, c = idx % DMODEL;
  int h = c / HD, d = c % HD;
  kT[((size_t)h * HD + d) * NTOK + n] = k[idx];
}

// ---------------------------------------------------------------------------
// Flash attention: 1 wave per (16-query tile, head); queries are latent rows
// only (qbase = LEN_X); keys/values span all NTOK tokens. Online softmax.
// ---------------------------------------------------------------------------
__global__ __launch_bounds__(32) void flash_attn_kernel(
    const bf16* __restrict__ q, const bf16* __restrict__ kT,
    const bf16* __restrict__ v, const int* __restrict__ ts,
    bf16* __restrict__ out, int qbase) {
  __shared__ bf16 Plds[16 * 32];
  int q0   = qbase + blockIdx.x * 16;
  int h    = blockIdx.y;
  int lane = threadIdx.x & 31;
  int half = lane >> 4;
  int cl   = lane & 15;
  const bf16* kTh = kT + (size_t)h * HD * NTOK;

  v16bf qf[4];
#pragma unroll
  for (int fc = 0; fc < 4; fc++)
    qf[fc] = load_a_frag(q + (size_t)q0 * DMODEL + h * HD + fc * 32, DMODEL);

  int tsq[8];
#pragma unroll
  for (int r = 0; r < 8; r++) tsq[r] = ts[q0 + (half ? 8 + r : r)];

  v8f o[8];
#pragma unroll
  for (int j = 0; j < 8; j++) o[j] = zero8();
  float mrow[8], lrow[8];
#pragma unroll
  for (int r = 0; r < 8; r++) { mrow[r] = -3.0e38f; lrow[r] = 0.0f; }
  const float scale = 0.0883883476483f;  // 1/sqrt(128)

  for (int k0 = 0; k0 < NTOK; k0 += 32) {
    int tsk0 = ts[k0 + cl];
    int tsk1 = ts[k0 + 16 + cl];
    v8f s0 = zero8(), s1 = zero8();
#pragma unroll
    for (int fc = 0; fc < 4; fc++) {
      v16bf b0 = load_b_frag(kTh + (size_t)(fc * 32) * NTOK + k0, NTOK);
      v16bf b1 = load_b_frag(kTh + (size_t)(fc * 32) * NTOK + k0 + 16, NTOK);
      s0 = wmma_bf16(qf[fc], b0, s0);
      s1 = wmma_bf16(qf[fc], b1, s1);
    }
#pragma unroll
    for (int r = 0; r < 8; r++) {
      float a0 = s0[r] * scale; if (tsq[r] < tsk0) a0 = -1.0e30f;
      float a1 = s1[r] * scale; if (tsq[r] < tsk1) a1 = -1.0e30f;
      float mx = fmaxf(a0, a1);
#pragma unroll
      for (int st = 1; st < 16; st <<= 1) mx = fmaxf(mx, __shfl_xor(mx, st, 32));
      float mnew = fmaxf(mrow[r], mx);
      float resc = __expf(mrow[r] - mnew);
      float p0   = __expf(a0 - mnew);
      float p1   = __expf(a1 - mnew);
      float psum = p0 + p1;
#pragma unroll
      for (int st = 1; st < 16; st <<= 1) psum += __shfl_xor(psum, st, 32);
      lrow[r] = lrow[r] * resc + psum;
      mrow[r] = mnew;
#pragma unroll
      for (int j = 0; j < 8; j++) o[j][r] *= resc;
      int Mr = (half ? 8 + r : r);
      Plds[Mr * 32 + cl]      = (bf16)p0;
      Plds[Mr * 32 + 16 + cl] = (bf16)p1;
    }
    __syncthreads();
    v16bf ap = load_a_frag(Plds, 32);
#pragma unroll
    for (int dc = 0; dc < 8; dc++) {
      v16bf bv = load_b_frag(v + (size_t)k0 * DMODEL + h * HD + dc * 16, DMODEL);
      o[dc] = wmma_bf16(ap, bv, o[dc]);
    }
    __syncthreads();
  }

#pragma unroll
  for (int r = 0; r < 8; r++) {
    float inv = 1.0f / lrow[r];
    int row   = q0 + (half ? 8 + r : r);
#pragma unroll
    for (int dc = 0; dc < 8; dc++)
      out[(size_t)row * DMODEL + h * HD + dc * 16 + cl] = (bf16)(o[dc][r] * inv);
  }
}

// ---------------------------------------------------------------------------
// Final readout: out[t,o] = dot(lat_flat[t], Wout[o]) + bout[o]. Pure BW.
// ---------------------------------------------------------------------------
__global__ __launch_bounds__(256) void final_out_kernel(
    const float* __restrict__ x, const float* __restrict__ Wout,
    const float* __restrict__ bout, float* __restrict__ out) {
  int t = blockIdx.y;
  int o = blockIdx.x;
  const float* a = x + (size_t)(LEN_X + t * SLAT) * DMODEL;  // contiguous 65536
  const float* w = Wout + (size_t)o * (SLAT * DMODEL);
  float s = 0.0f;
  for (int i = threadIdx.x * 4; i < SLAT * DMODEL; i += 256 * 4) {
    float4 av = *(const float4*)(a + i);
    float4 wv = *(const float4*)(w + i);
    s += av.x * wv.x + av.y * wv.y + av.z * wv.z + av.w * wv.w;
  }
  __shared__ float red[256];
  red[threadIdx.x] = s;
  __syncthreads();
  for (int st = 128; st > 0; st >>= 1) {
    if (threadIdx.x < st) red[threadIdx.x] += red[threadIdx.x + st];
    __syncthreads();
  }
  if (threadIdx.x == 0) out[t * OUTDIM + o] = red[0] + bout[o];
}

// ---------------------------------------------------------------------------
extern "C" void kernel_launch(void* const* d_in, const int* in_sizes, int n_in,
                              void* d_out, int out_size, void* d_ws,
                              size_t ws_size, hipStream_t stream) {
  (void)in_sizes; (void)n_in; (void)out_size; (void)ws_size;
  const float* data   = (const float*)d_in[0];
  const int*   tsteps = (const int*)d_in[1];
  const float* latent = (const float*)d_in[2];
  const float* Wq  = (const float*)d_in[3];
  const float* Wk  = (const float*)d_in[4];
  const float* Wv  = (const float*)d_in[5];
  const float* Wo  = (const float*)d_in[6];
  const float* G1  = (const float*)d_in[7];
  const float* B1  = (const float*)d_in[8];
  const float* G2  = (const float*)d_in[9];
  const float* B2  = (const float*)d_in[10];
  const float* W1  = (const float*)d_in[11];
  const float* Bm1 = (const float*)d_in[12];
  const float* W2  = (const float*)d_in[13];
  const float* Bm2 = (const float*)d_in[14];
  const float* Wout = (const float*)d_in[15];
  const float* bout = (const float*)d_in[16];
  float* out = (float*)d_out;

  char*  ws  = (char*)d_ws;
  size_t off = 0;
  auto alloc = [&](size_t bytes) -> char* {
    char* p = ws + off;
    off += (bytes + 255) & ~(size_t)255;
    return p;
  };
  float* x    = (float*)alloc((size_t)NTOK * DMODEL * 4);
  int*   ts   = (int*)alloc((size_t)NTOK * 4);
  bf16* hbuf  = (bf16*)alloc((size_t)NTOK * DMODEL * 2);
  bf16* qbuf  = (bf16*)alloc((size_t)NTOK * DMODEL * 2);
  bf16* kbuf  = (bf16*)alloc((size_t)NTOK * DMODEL * 2);
  bf16* vbuf  = (bf16*)alloc((size_t)NTOK * DMODEL * 2);
  bf16* kT    = (bf16*)alloc((size_t)NHEAD * HD * NTOK * 2);
  bf16* attnb = (bf16*)alloc((size_t)NTOK * DMODEL * 2);
  bf16* ffn   = (bf16*)alloc((size_t)NLAT * FFDIM * 2);
  bf16* wq_s  = (bf16*)alloc((size_t)DMODEL * DMODEL * 2);
  bf16* wk_s  = (bf16*)alloc((size_t)DMODEL * DMODEL * 2);
  bf16* wv_s  = (bf16*)alloc((size_t)DMODEL * DMODEL * 2);
  bf16* wo_s  = (bf16*)alloc((size_t)DMODEL * DMODEL * 2);
  bf16* w1_s  = (bf16*)alloc((size_t)DMODEL * FFDIM * 2);
  bf16* w2_s  = (bf16*)alloc((size_t)FFDIM * DMODEL * 2);

  build_x_kernel<<<(NTOK * DMODEL + 255) / 256, 256, 0, stream>>>(
      data, tsteps, latent, x, ts);

  const int nDD = DMODEL * DMODEL;
  const int nDF = DMODEL * FFDIM;
  for (int l = 0; l < NLAYER; l++) {
    cvt_bf16_kernel<<<(nDD + 255) / 256, 256, 0, stream>>>(Wq + (size_t)l * nDD, wq_s, nDD);
    cvt_bf16_kernel<<<(nDD + 255) / 256, 256, 0, stream>>>(Wk + (size_t)l * nDD, wk_s, nDD);
    cvt_bf16_kernel<<<(nDD + 255) / 256, 256, 0, stream>>>(Wv + (size_t)l * nDD, wv_s, nDD);
    cvt_bf16_kernel<<<(nDD + 255) / 256, 256, 0, stream>>>(Wo + (size_t)l * nDD, wo_s, nDD);
    cvt_bf16_kernel<<<(nDF + 255) / 256, 256, 0, stream>>>(W1 + (size_t)l * nDF, w1_s, nDF);
    cvt_bf16_kernel<<<(nDF + 255) / 256, 256, 0, stream>>>(W2 + (size_t)l * nDF, w2_s, nDF);

    // LN1 + Q/K/V over ALL tokens (context rows feed K/V)
    layernorm_kernel<<<NTOK, 256, 0, stream>>>(x, G1 + l * DMODEL, B1 + l * DMODEL, hbuf);
    dim3 gQKV(DMODEL / 128, NTOK / 256);
    gemm_kernel<<<gQKV, 256, 0, stream>>>(hbuf, wq_s, nullptr, qbuf, nullptr, NTOK, DMODEL, DMODEL, 0);
    gemm_kernel<<<gQKV, 256, 0, stream>>>(hbuf, wk_s, nullptr, kbuf, nullptr, NTOK, DMODEL, DMODEL, 0);
    gemm_kernel<<<gQKV, 256, 0, stream>>>(hbuf, wv_s, nullptr, vbuf, nullptr, NTOK, DMODEL, DMODEL, 0);

    transpose_k_kernel<<<(NTOK * DMODEL + 255) / 256, 256, 0, stream>>>(kbuf, kT);

    // attention for latent queries only; outputs for context rows are discarded
    dim3 gA(NLAT / 16, NHEAD);
    flash_attn_kernel<<<gA, 32, 0, stream>>>(qbuf, kT, vbuf, ts, attnb, LEN_X);

    // residual O-projection, LN2, FFN: latent rows only
    dim3 gDDl(DMODEL / 128, NLAT / 256);
    gemm_kernel<<<gDDl, 256, 0, stream>>>(attnb + (size_t)LEN_X * DMODEL, wo_s,
                                          nullptr, nullptr, x + (size_t)LEN_X * DMODEL,
                                          NLAT, DMODEL, DMODEL, 2);

    layernorm_kernel<<<NLAT, 256, 0, stream>>>(x + (size_t)LEN_X * DMODEL,
                                               G2 + l * DMODEL, B2 + l * DMODEL, hbuf);

    dim3 gF1(FFDIM / 128, NLAT / 256);
    gemm_kernel<<<gF1, 256, 0, stream>>>(hbuf, w1_s, Bm1 + (size_t)l * FFDIM, ffn,
                                         nullptr, NLAT, FFDIM, DMODEL, 1);
    gemm_kernel<<<gDDl, 256, 0, stream>>>(ffn, w2_s, Bm2 + (size_t)l * DMODEL,
                                          nullptr, x + (size_t)LEN_X * DMODEL,
                                          NLAT, DMODEL, FFDIM, 2);
  }

  dim3 gf(OUTDIM, TSTEP);
  final_out_kernel<<<gf, 256, 0, stream>>>(x, Wout, bout, out);
}
